// GNN_node_57166014710233
// MI455X (gfx1250) — matrix operand order repaired and checked by reading, not compiled, not run on hardware
//
#include <hip/hip_runtime.h>

#define D_DIM 256
#define F_DIM 128
#define H1_DIM 512
#define L_NUM 5
#define BROW 40   // padded LDS B row stride in bf16 (80B) -> conflict-free ds_load_b128

typedef __attribute__((ext_vector_type(16))) __bf16 bf16x16;
typedef __attribute__((ext_vector_type(8)))  __bf16 bf16x8;
typedef __attribute__((ext_vector_type(8)))  float  f32x8;

union ABFrag { bf16x16 v; bf16x8 h[2]; };

__device__ __forceinline__ __bf16 f2bf(float f) {
  unsigned u = __builtin_bit_cast(unsigned, f);
  u += 0x7FFFu + ((u >> 16) & 1u);            // round-to-nearest-even
  unsigned short s = (unsigned short)(u >> 16);
  return __builtin_bit_cast(__bf16, s);
}

// ---------------- elementwise helpers ----------------

__global__ void cvt_f32_to_bf16(const float* __restrict__ src, __bf16* __restrict__ dst, int n) {
  int i = blockIdx.x * blockDim.x + threadIdx.x;
  if (i < n) dst[i] = f2bf(src[i]);
}

__global__ void zero_f32(float* __restrict__ p, int n) {
  int i = blockIdx.x * blockDim.x + threadIdx.x;
  if (i < n) p[i] = 0.0f;
}

// ---------------- edge scatter: agg[dst] += relu(h[src]) ----------------

__global__ __launch_bounds__(256) void scatter_relu_add(
    const float* __restrict__ h, const int* __restrict__ src, const int* __restrict__ dst,
    float* __restrict__ agg, int E) {
  int e = blockIdx.x * 4 + (threadIdx.x >> 6);
  if (e >= E) return;
  int c = (threadIdx.x & 63) * 4;
  int s = src[e], d = dst[e];
  const float4 v = *(const float4*)(h + (size_t)s * D_DIM + c);
  float* pa = agg + (size_t)d * D_DIM + c;
  atomicAdd(pa + 0, fmaxf(v.x, 0.0f));
  atomicAdd(pa + 1, fmaxf(v.y, 0.0f));
  atomicAdd(pa + 2, fmaxf(v.z, 0.0f));
  atomicAdd(pa + 3, fmaxf(v.w, 0.0f));
}

// ---------------- fragment packers ----------------

__device__ __forceinline__ void pack4_eps(bf16x8& dv, int o, float4 hv, float4 gv, float e) {
  dv[o + 0] = f2bf(fmaf(e, hv.x, gv.x));
  dv[o + 1] = f2bf(fmaf(e, hv.y, gv.y));
  dv[o + 2] = f2bf(fmaf(e, hv.z, gv.z));
  dv[o + 3] = f2bf(fmaf(e, hv.w, gv.w));
}

__device__ __forceinline__ void pack4_bnrelu(bf16x8& dv, int o, float4 z, float4 s, float4 b) {
  dv[o + 0] = f2bf(fmaxf(fmaf(z.x, s.x, b.x), 0.0f));
  dv[o + 1] = f2bf(fmaxf(fmaf(z.y, s.y, b.y), 0.0f));
  dv[o + 2] = f2bf(fmaxf(fmaf(z.z, s.z, b.z), 0.0f));
  dv[o + 3] = f2bf(fmaxf(fmaf(z.w, s.w, b.w), 0.0f));
}

// ---------------- WMMA GEMM: C[M,N] = A[M,K] @ W[N,K]^T (+bias, +stats) ----------------
// MODE 0: A = precomputed bf16 (encoder). No bias/stats.
// MODE 1: A = (1+eps)*h + agg (fp32 on the fly). bias=b1, stats -> gsum/gsq.
// MODE 2: A = relu(z1*kscale[k] + kbias[k]) (fused BN1+ReLU). bias=b2, stats.
// Block: 256 thr = 8 waves, 2(M)x4(N); wave tile 16x64; K step 32.
// B tile (256 rows x 32 K bf16) staged to LDS via async global->LDS copies
// (ASYNCcnt), double-buffered; steady-state loop is branch-free (last k-step
// peeled). LDS rows padded to 80B so ds_load_b128 fragment reads are
// bank-conflict free (bank start = row*20 mod 64 covers all banks).

template<int MODE>
__global__ __launch_bounds__(256, 2) void gemm_wmma(
    const __bf16* __restrict__ Abf,
    const float* __restrict__ Af0, const float* __restrict__ Af1,
    const float* __restrict__ kscale, const float* __restrict__ kbias,
    const float* __restrict__ epsp,
    const __bf16* __restrict__ W, const float* __restrict__ nbias,
    float* __restrict__ C, float* __restrict__ gsum, float* __restrict__ gsq,
    int M, int K, int ldc)
{
  __shared__ float s_sum[256];
  __shared__ float s_sq[256];
  __shared__ __bf16 sB[2][256][BROW];   // double-buffered padded B tile, 2 x 20KB

  const int tid  = threadIdx.x;
  const int wave = tid >> 5;
  const int lane = tid & 31;
  const int wm   = wave >> 2;       // 0..1 : M sub-tile
  const int wn   = wave & 3;        // 0..3 : N sub-tile (64 cols)
  const int l    = lane & 15;
  const int hi   = lane >> 4;
  const int m0   = blockIdx.x * 32 + wm * 16;
  const int nblkbase = blockIdx.y * 256;

  if (MODE != 0) { s_sum[tid] = 0.0f; s_sq[tid] = 0.0f; }

  int am = m0 + l; if (am > M - 1) am = M - 1;   // clamp A row (stores are guarded)
  float epsv = 1.0f;
  if (MODE == 1) epsv += epsp[0];

  // Async-stage one 256x32 B tile: thread t copies 64B row (nblkbase+t, k0..k0+31).
  // The offset: immediate is added to both the global address and the LDS
  // address (VDST) per the VGLOBAL async addressing rules (row-local offsets).
  auto stageB = [&](int k0, int buf) {
    unsigned ldsa = (unsigned)(size_t)(void*)&sB[buf][tid][0];
    const __bf16* gp = W + (size_t)(nblkbase + tid) * K + k0;
    asm volatile(
        "global_load_async_to_lds_b128 %0, %1, off\n\t"
        "global_load_async_to_lds_b128 %0, %1, off offset:16\n\t"
        "global_load_async_to_lds_b128 %0, %1, off offset:32\n\t"
        "global_load_async_to_lds_b128 %0, %1, off offset:48"
        :: "v"(ldsa), "v"(gp) : "memory");
  };

  // A-fragment builder (MODE-specific, constant-folded).
  auto loadA = [&](ABFrag& a, int k0) {
    if (MODE == 0) {
      const __bf16* pa = Abf + (size_t)am * K + k0 + hi * 8;
      a.h[0] = *(const bf16x8*)pa;          // K = hk..hk+7
      a.h[1] = *(const bf16x8*)(pa + 16);   // K = hk+16..hk+23
    } else if (MODE == 1) {
      const size_t base = (size_t)am * K + k0 + hi * 8;
      const float* ph = Af0 + base;
      const float* pg = Af1 + base;
      pack4_eps(a.h[0], 0, *(const float4*)(ph),      *(const float4*)(pg),      epsv);
      pack4_eps(a.h[0], 4, *(const float4*)(ph + 4),  *(const float4*)(pg + 4),  epsv);
      pack4_eps(a.h[1], 0, *(const float4*)(ph + 16), *(const float4*)(pg + 16), epsv);
      pack4_eps(a.h[1], 4, *(const float4*)(ph + 20), *(const float4*)(pg + 20), epsv);
    } else {
      const size_t base = (size_t)am * K + k0 + hi * 8;
      const float* pz = Af0 + base;
      const float* ps = kscale + k0 + hi * 8;
      const float* pb = kbias  + k0 + hi * 8;
      pack4_bnrelu(a.h[0], 0, *(const float4*)(pz),      *(const float4*)(ps),      *(const float4*)(pb));
      pack4_bnrelu(a.h[0], 4, *(const float4*)(pz + 4),  *(const float4*)(ps + 4),  *(const float4*)(pb + 4));
      pack4_bnrelu(a.h[1], 0, *(const float4*)(pz + 16), *(const float4*)(ps + 16), *(const float4*)(pb + 16));
      pack4_bnrelu(a.h[1], 4, *(const float4*)(pz + 20), *(const float4*)(ps + 20), *(const float4*)(pb + 20));
    }
  };

  f32x8 c[4];
  #pragma unroll
  for (int i = 0; i < 4; ++i)
    #pragma unroll
    for (int j = 0; j < 8; ++j) c[i][j] = 0.0f;

  stageB(0, 0);
  int cur = 0;
  int k0 = 0;

  // Steady state: branch-free (always a next tile in flight).
  for (; k0 + 32 < K; k0 += 32) {
    stageB(k0 + 32, cur ^ 1);

    ABFrag a;
    loadA(a, k0);                       // issued before the async drain

    // In-order async completion: <=4 outstanding leaves only the next tile's
    // copies in flight -> current tile is resident; barrier makes it visible.
    asm volatile("s_wait_asynccnt 4" ::: "memory");
    __syncthreads();

    bf16x16 bf[4];
    #pragma unroll
    for (int cs = 0; cs < 4; ++cs)
      bf[cs] = __builtin_shufflevector(*(const bf16x8*)&sB[cur][wn * 64 + cs * 16 + l][hi * 8], *(const bf16x8*)&sB[cur][wn * 64 + cs * 16 + l][16 + hi * 8], 0, 1, 2, 3, 4, 5, 6, 7, 8, 9, 10, 11, 12, 13, 14, 15);
    #pragma unroll
    for (int cs = 0; cs < 4; ++cs)
      c[cs] = __builtin_amdgcn_wmma_f32_16x16x32_bf16(
          false, a.v, false, bf[cs], (short)0, c[cs], false, false);

    __syncthreads();   // all waves done reading buf[cur] before it is re-filled
    cur ^= 1;
  }

  // Peeled tail k-step.
  {
    ABFrag a;
    loadA(a, k0);
    asm volatile("s_wait_asynccnt 0" ::: "memory");
    __syncthreads();
    bf16x16 bf[4];
    #pragma unroll
    for (int cs = 0; cs < 4; ++cs)
      bf[cs] = __builtin_shufflevector(*(const bf16x8*)&sB[cur][wn * 64 + cs * 16 + l][hi * 8], *(const bf16x8*)&sB[cur][wn * 64 + cs * 16 + l][16 + hi * 8], 0, 1, 2, 3, 4, 5, 6, 7, 8, 9, 10, 11, 12, 13, 14, 15);
    #pragma unroll
    for (int cs = 0; cs < 4; ++cs)
      c[cs] = __builtin_amdgcn_wmma_f32_16x16x32_bf16(
          false, a.v, false, bf[cs], (short)0, c[cs], false, false);
  }

  // Epilogue: bias, store, fused column stats for BatchNorm.
  #pragma unroll
  for (int cs = 0; cs < 4; ++cs) {
    const int col = nblkbase + wn * 64 + cs * 16 + l;
    const float bv = (MODE != 0) ? nbias[col] : 0.0f;
    float psum = 0.0f, psq = 0.0f;
    #pragma unroll
    for (int r = 0; r < 8; ++r) {
      const int row = m0 + hi * 8 + r;
      float v = c[cs][r] + bv;
      if (row < M) {
        C[(size_t)row * ldc + col] = v;
        psum += v; psq += v * v;
      }
    }
    if (MODE != 0) {
      const int rel = wn * 64 + cs * 16 + l;
      atomicAdd(&s_sum[rel], psum);
      atomicAdd(&s_sq[rel], psq);
    }
  }
  if (MODE != 0) {
    __syncthreads();
    atomicAdd(&gsum[nblkbase + tid], s_sum[tid]);
    atomicAdd(&gsq [nblkbase + tid], s_sq[tid]);
  }
}

// ---------------- BN stats -> affine (scale, bias) ----------------

__global__ void bn_affine(const float* __restrict__ sum, const float* __restrict__ sq,
                          const float* __restrict__ g, const float* __restrict__ b,
                          float* __restrict__ scale, float* __restrict__ bias,
                          int Cn, float invN) {
  int i = blockIdx.x * blockDim.x + threadIdx.x;
  if (i >= Cn) return;
  float mu  = sum[i] * invN;
  float var = fmaxf(sq[i] * invN - mu * mu, 0.0f);
  float sc  = g[i] * rsqrtf(var + 1e-5f);
  scale[i] = sc;
  bias[i]  = b[i] - mu * sc;
}

// ---------------- finalize: h = bn(z2) (+relu), optional extra output copy ----------------

__global__ void finalize_bn(const float* __restrict__ z2, const float* __restrict__ scale,
                            const float* __restrict__ bias, float* __restrict__ h,
                            float* __restrict__ extra, int total, int relu) {
  int i = blockIdx.x * blockDim.x + threadIdx.x;
  if (i >= total) return;
  int col = i & (D_DIM - 1);
  float v = fmaf(z2[i], scale[col], bias[col]);
  if (relu) v = fmaxf(v, 0.0f);
  h[i] = v;
  if (extra) extra[i] = v;
}

// ---------------- host orchestration ----------------

extern "C" void kernel_launch(void* const* d_in, const int* in_sizes, int n_in,
                              void* d_out, int out_size, void* d_ws, size_t ws_size,
                              hipStream_t stream) {
  const float* x    = (const float*)d_in[0];
  const int*   ei   = (const int*)d_in[1];
  const float* fcw  = (const float*)d_in[2];
  const float* eps  = (const float*)d_in[3];
  const float* w1   = (const float*)d_in[4];
  const float* b1   = (const float*)d_in[5];
  const float* bn1g = (const float*)d_in[6];
  const float* bn1b = (const float*)d_in[7];
  const float* w2   = (const float*)d_in[8];
  const float* b2   = (const float*)d_in[9];
  const float* bng  = (const float*)d_in[10];
  const float* bnb  = (const float*)d_in[11];

  const int N = in_sizes[0] / F_DIM;
  const int E = in_sizes[1] / 2;
  float* out = (float*)d_out;

  char* ws = (char*)d_ws;
  size_t off = 0;
  auto alloc = [&](size_t bytes) -> char* {
    char* p = ws + off;
    off = (off + bytes + 255) & ~(size_t)255;
    return p;
  };

  __bf16* x_bf   = (__bf16*)alloc((size_t)N * F_DIM * 2);
  __bf16* fcw_bf = (__bf16*)alloc((size_t)D_DIM * F_DIM * 2);
  __bf16* w1_bf  = (__bf16*)alloc((size_t)L_NUM * H1_DIM * D_DIM * 2);
  __bf16* w2_bf  = (__bf16*)alloc((size_t)L_NUM * D_DIM * H1_DIM * 2);
  float* h    = (float*)alloc((size_t)N * D_DIM * 4);
  float* agg  = (float*)alloc((size_t)N * D_DIM * 4);
  float* z1   = (float*)alloc((size_t)N * H1_DIM * 4);
  float* z2   = (float*)alloc((size_t)N * D_DIM * 4);
  float* sum1 = (float*)alloc(H1_DIM * 4);   // stats block: contiguous (all sizes 256B-mult)
  float* sq1  = (float*)alloc(H1_DIM * 4);
  float* sum2 = (float*)alloc(D_DIM * 4);
  float* sq2  = (float*)alloc(D_DIM * 4);
  float* sc1  = (float*)alloc(H1_DIM * 4);
  float* bi1  = (float*)alloc(H1_DIM * 4);
  float* sc2  = (float*)alloc(D_DIM * 4);
  float* bi2  = (float*)alloc(D_DIM * 4);

  auto cdiv = [](int a, int b) { return (a + b - 1) / b; };

  cvt_f32_to_bf16<<<cdiv(N * F_DIM, 256), 256, 0, stream>>>(x, x_bf, N * F_DIM);
  cvt_f32_to_bf16<<<cdiv(D_DIM * F_DIM, 256), 256, 0, stream>>>(fcw, fcw_bf, D_DIM * F_DIM);
  cvt_f32_to_bf16<<<cdiv(L_NUM * H1_DIM * D_DIM, 256), 256, 0, stream>>>(w1, w1_bf, L_NUM * H1_DIM * D_DIM);
  cvt_f32_to_bf16<<<cdiv(L_NUM * D_DIM * H1_DIM, 256), 256, 0, stream>>>(w2, w2_bf, L_NUM * D_DIM * H1_DIM);

  const int gx = cdiv(N, 32);

  // Encoder: h = x @ fc_w^T   [N,128]x[256,128]^T
  gemm_wmma<0><<<dim3(gx, 1), 256, 0, stream>>>(
      x_bf, nullptr, nullptr, nullptr, nullptr, nullptr,
      fcw_bf, nullptr, h, nullptr, nullptr, N, F_DIM, D_DIM);

  for (int lyr = 0; lyr < L_NUM; ++lyr) {
    zero_f32<<<cdiv(N * D_DIM, 256), 256, 0, stream>>>(agg, N * D_DIM);
    zero_f32<<<cdiv(2 * (H1_DIM + D_DIM), 256), 256, 0, stream>>>(sum1, 2 * (H1_DIM + D_DIM));

    scatter_relu_add<<<cdiv(E, 4), 256, 0, stream>>>(h, ei, ei + E, agg, E);

    // z1 = ((1+eps)h + agg) @ w1^T + b1 ; fused column stats
    gemm_wmma<1><<<dim3(gx, H1_DIM / 256), 256, 0, stream>>>(
        nullptr, h, agg, nullptr, nullptr, eps + lyr,
        w1_bf + (size_t)lyr * H1_DIM * D_DIM, b1 + (size_t)lyr * H1_DIM,
        z1, sum1, sq1, N, D_DIM, H1_DIM);

    bn_affine<<<cdiv(H1_DIM, 256), 256, 0, stream>>>(
        sum1, sq1, bn1g + (size_t)lyr * H1_DIM, bn1b + (size_t)lyr * H1_DIM,
        sc1, bi1, H1_DIM, 1.0f / (float)N);

    // z2 = relu(bn1(z1)) @ w2^T + b2 ; fused column stats
    gemm_wmma<2><<<dim3(gx, 1), 256, 0, stream>>>(
        nullptr, z1, nullptr, sc1, bi1, nullptr,
        w2_bf + (size_t)lyr * D_DIM * H1_DIM, b2 + (size_t)lyr * D_DIM,
        z2, sum2, sq2, N, H1_DIM, D_DIM);

    bn_affine<<<1, 256, 0, stream>>>(
        sum2, sq2, bng + (size_t)lyr * D_DIM, bnb + (size_t)lyr * D_DIM,
        sc2, bi2, D_DIM, 1.0f / (float)N);

    float* extra = (lyr == 0) ? (out + (size_t)N * D_DIM)
                 : (lyr == L_NUM - 1) ? out : nullptr;
    finalize_bn<<<cdiv(N * D_DIM, 256), 256, 0, stream>>>(
        z2, sc2, bi2, h, extra, N * D_DIM, (lyr < L_NUM - 1) ? 1 : 0);
  }
}
